// DARNN_45784351375644
// MI455X (gfx1250) — matrix operand order repaired and checked by reading, not compiled, not run on hardware
//
#include <hip/hip_runtime.h>
#include <hip/hip_bf16.h>
#include <stdint.h>

// ---------------------------------------------------------------------------
// DA-RNN for MI455X (gfx1250): one workgroup per 16-row batch tile, full
// 255-step recurrence per WG, all dense projections via v_wmma_f32_16x16x32_bf16.
// A operands kept in bf16 WMMA-fragment order in LDS; the strided x[:, :, t]
// column gather is delegated to the Tensor Data Mover (tensor_load_to_lds).
// ---------------------------------------------------------------------------

typedef __attribute__((ext_vector_type(16))) __bf16 v16bf;
typedef __attribute__((ext_vector_type(8)))  float  v8f;
typedef __attribute__((ext_vector_type(4)))  unsigned int u32x4;
typedef __attribute__((ext_vector_type(8)))  int  i32x8;
typedef __attribute__((ext_vector_type(4)))  int  i32x4;

#if defined(__gfx1250__) && __has_builtin(__builtin_amdgcn_tensor_load_to_lds) && \
    __has_builtin(__builtin_amdgcn_s_wait_tensorcnt)
#define HAVE_TDM 1
#else
#define HAVE_TDM 0
#endif

#define B_SZ 1024
#define NV   96
#define W_SZ 128
#define E_SZ 64
#define D_SZ 64

#define HC_LD   132   // fp32 [h(64)|c(64)] state, +4 pad to spread LDS banks
#define WOUT_LD 132
#define GATE_LD 260

union AFrag { v16bf v; __bf16 e[16]; };

__device__ __forceinline__ v8f wmma_bf16(v16bf a, v16bf b, v8f c) {
  // D(16x16,f32) = A(16x32,bf16) x B(32x16,bf16) + C
  return __builtin_amdgcn_wmma_f32_16x16x32_bf16(false, a, false, b, (short)0, c,
                                                 false, false);
}

// Load a 16x32 A-fragment from row-major f32 (global), convert to bf16.
// ISA layout (16-bit A 16x32): lane m=L&15, kh=L>>4;
//   elems 0..7  -> K = kbase + kh*8 + j ; elems 8..15 -> K = kbase+16+kh*8+j
__device__ __forceinline__ v16bf loadA_f32(const float* src, int ld, int kbase,
                                           int lane) {
  AFrag a;
  const int m = lane & 15, kh = lane >> 4;
  const float* r0 = src + m * ld + kbase + kh * 8;
  const float* r1 = r0 + 16;
#pragma unroll
  for (int j = 0; j < 8; ++j) a.e[j] = (__bf16)r0[j];
#pragma unroll
  for (int j = 0; j < 8; ++j) a.e[8 + j] = (__bf16)r1[j];
  return a.v;
}

// Scatter one (row m, col k) element into a bf16 A-fragment-ordered buffer.
__device__ __forceinline__ void packA_elem(__bf16* buf, int m, int col, float v) {
  const int kt = col >> 5, kin = col & 31;
  const int kh = (kin >> 3) & 1;
  const int j  = (kin & 7) + ((kin & 16) ? 8 : 0);
  buf[kt * 512 + (((kh << 4) + m) << 4) + j] = (__bf16)v;
}

// Store C/D fragment (f32 16x16) + per-column bias into row-major f32.
__device__ __forceinline__ void storeC_bias(float* dst, int ld, v8f c,
                                            const float* bias, int lane) {
  const int n = lane & 15, mh = lane >> 4;
  const float bv = bias[n];
#pragma unroll
  for (int r = 0; r < 8; ++r) dst[(mh * 8 + r) * ld + n] = c[r] + bv;
}

__device__ __forceinline__ float fsig(float x) {
  return 1.f / (1.f + __expf(-x));
}
__device__ __forceinline__ float ftanh(float x) {
  float e = __expf(2.f * x);
  return 1.f - 2.f / (e + 1.f);
}

__device__ __forceinline__ float wave_sum(float v) {
#pragma unroll
  for (int o = 16; o > 0; o >>= 1) v += __shfl_xor(v, o);
  return v;
}
__device__ __forceinline__ float wave_max(float v) {
#pragma unroll
  for (int o = 16; o > 0; o >>= 1) v = fmaxf(v, __shfl_xor(v, o));
  return v;
}

// ---------------------------------------------------------------------------
// Pack weight (stored as N rows x K cols, B[k][n] = src[n*K+k]) into WMMA
// B-fragment order: tile (kt,nt) -> 32 lanes x 16 bf16 per lane, contiguous.
// ---------------------------------------------------------------------------
__global__ __launch_bounds__(32) void pack_b(const float* __restrict__ src,
                                             __bf16* __restrict__ dst,
                                             int K, int N) {
  const int NT = N >> 4;
  const int tile = blockIdx.x;
  const int kt = tile / NT, nt = tile - kt * NT;
  const int lane = threadIdx.x;
  const int n  = nt * 16 + (lane & 15);
  const int kb = kt * 32 + (lane >> 4) * 8;
  __bf16* o = dst + (size_t)tile * 512 + (size_t)lane * 16;
  const float* s = src + (size_t)n * K + kb;
#pragma unroll
  for (int j = 0; j < 16; ++j) o[j] = (__bf16)s[(j & 7) + ((j & 8) << 1)];
}

// ---------------------------------------------------------------------------
// u_e[b,n,:] = x[b,n,:] @ enc_lu^T + enc_lub   (precompute; L2-resident 50MB)
// ---------------------------------------------------------------------------
__global__ __launch_bounds__(256) void ue_kernel(const float* __restrict__ x,
                                                 const __bf16* __restrict__ Blu_,
                                                 const float* __restrict__ lub,
                                                 float* __restrict__ ue) {
  const int b  = blockIdx.x;
  const int mt = blockIdx.y;
  const int lane = threadIdx.x & 31, wv = threadIdx.x >> 5;
  const v16bf* Blu = (const v16bf*)Blu_;
  const float* A = x + ((size_t)b * NV + mt * 16) * W_SZ;  // 16 rows, ld=128
  v8f acc = {0.f, 0.f, 0.f, 0.f, 0.f, 0.f, 0.f, 0.f};
#pragma unroll
  for (int kt = 0; kt < 4; ++kt) {
    v16bf a  = loadA_f32(A, W_SZ, kt * 32, lane);
    v16bf bb = Blu[(kt * 8 + wv) * 32 + lane];
    acc = wmma_bf16(a, bb, acc);
  }
  const int n = lane & 15, mh = lane >> 4;
  const float bv = lub[wv * 16 + n];
#pragma unroll
  for (int r = 0; r < 8; ++r) {
    const int m = mh * 8 + r;
    ue[((size_t)b * NV + mt * 16 + m) * W_SZ + wv * 16 + n] = acc[r] + bv;
  }
}

// ---------------------------------------------------------------------------
// Main persistent kernel: 64 WGs x 256 threads, 16 batch rows each.
// ---------------------------------------------------------------------------
struct DP {
  const float *x, *enc_lv, *enc_lwb, *enc_bih, *enc_bhh;
  const float *dec_Wih, *dec_bih, *dec_bhh, *dec_lv, *dec_lwb, *dec_lub;
  const float *dec_lp, *dec_lpb, *dec_ly, *dec_lyb;
  const int* target;
  const __bf16 *Blw, *BWih, *BWhh, *Bdlw, *Bdlu, *Bdwhh;
  const float* ue;
  float *hseq, *ud, *out;
};

__global__ __launch_bounds__(256) void darnn_main(DP p) {
  __shared__ float sh_hc[16 * HC_LD];     // fp32 [h | c] state
  __shared__ float sh_wout[16 * WOUT_LD];
  __shared__ float sh_g[16 * GATE_LD];
  __shared__ float sh_sc[2048];           // enc: 16x96 scores; dec: 128x16 v/beta
  __shared__ float sh_ctx[1024];
  __shared__ float sh_lv[128], sh_lwb[128], sh_b256e[256];
  __shared__ float sh_lvd[64], sh_lwbd[64], sh_b256d[256];
  __shared__ float sh_yt[16];
  __shared__ __align__(16) float sh_xcol[16 * NV];  // TDM-staged x[:, :, t]
  // bf16 A operands in WMMA fragment order (loaded as ds_load_b128 pairs)
  __shared__ __align__(32) __bf16 sh_hcA[4 * 512];  // [h|c] 16x128, 4 k-tiles
  __shared__ __align__(32) __bf16 sh_xtA[3 * 512];  // x_tilde 16x96, 3 k-tiles

  const int tid  = threadIdx.x;
  const int lane = tid & 31, wv = tid >> 5;
  const int bg = blockIdx.x;
  const int b0 = bg * 16;

  if (tid < 128) { sh_lv[tid] = p.enc_lv[tid]; sh_lwb[tid] = p.enc_lwb[tid]; }
  sh_b256e[tid] = p.enc_bih[tid] + p.enc_bhh[tid];
  sh_b256d[tid] = p.dec_bih[tid] + p.dec_bhh[tid];
  if (tid < 64) { sh_lvd[tid] = p.dec_lv[tid]; sh_lwbd[tid] = p.dec_lwb[tid]; }
  for (int i = tid; i < 16 * HC_LD; i += 256) sh_hc[i] = 0.f;
  for (int i = tid; i < 4 * 512; i += 256) sh_hcA[i] = (__bf16)0.f;
  __syncthreads();

  const v16bf* Blw   = (const v16bf*)p.Blw;
  const v16bf* BWih  = (const v16bf*)p.BWih;
  const v16bf* BWhh  = (const v16bf*)p.BWhh;
  const v16bf* Bdlw  = (const v16bf*)p.Bdlw;
  const v16bf* Bdlu  = (const v16bf*)p.Bdlu;
  const v16bf* Bdwhh = (const v16bf*)p.Bdwhh;
  const v16bf* hcA   = (const v16bf*)sh_hcA;
  const v16bf* xtA   = (const v16bf*)sh_xtA;

  // ======================= encoder: 128 steps ==============================
  for (int t = 0; t < W_SZ; ++t) {
#if HAVE_TDM
    // Kick the TDM: gather column x[b0:b0+16, :, t] (1536 elems, stride 128)
    // into sh_xcol as a 2-D tile (tile_dim0 = 1 elem, tile_dim1 = 1536 rows).
    // Latency hidden behind the w_out GEMM + scores/tanh phases below.
    if (wv == 0) {
      const uint64_t ga = (uint64_t)(p.x + ((size_t)b0 * NV) * W_SZ + t);
      const unsigned lds = (unsigned)(uintptr_t)sh_xcol;
      u32x4 g0 = {0u, 0u, 0u, 0u};
      g0[0] = 1u;                                   // count=1 (valid user D#)
      g0[1] = lds;                                  // lds_addr
      g0[2] = (unsigned)ga;                         // global_addr[31:0]
      g0[3] = (unsigned)((ga >> 32) & 0x1FFFFFFu)   // global_addr[56:32]
              | (2u << 30);                         // type=2 ("image")
      i32x8 g1 = {0, 0, 0, 0, 0, 0, 0, 0};
      g1[0] = (int)(2u << 16);                      // data_size=4B
      g1[1] = (int)((W_SZ & 0xFFFFu) << 16);        // tensor_dim0 = 128 (lo16)
      g1[2] = (int)((16 * NV & 0xFFFF) << 16);      // dim0 hi16=0 | tensor_dim1 lo16
      g1[3] = (int)(1u << 16);                      // tensor_dim1 hi16=0 | tile_dim0=1
      g1[4] = (int)(16 * NV);                       // tile_dim1=1536, tile_dim2=0
      g1[5] = (int)W_SZ;                            // tensor_dim0_stride = 128 (lo32)
      i32x4 z4 = {0, 0, 0, 0};
      i32x8 z8 = {0, 0, 0, 0, 0, 0, 0, 0};
      __builtin_amdgcn_tensor_load_to_lds(g0, g1, z4, z4, z8, 0);
    }
#endif

    // w_out(16x128) = [h,c](16x128) @ enc_lw^T + lwb ; 8 N-tiles, one per wave
    {
      v8f acc = {0.f, 0.f, 0.f, 0.f, 0.f, 0.f, 0.f, 0.f};
#pragma unroll
      for (int kt = 0; kt < 4; ++kt)
        acc = wmma_bf16(hcA[kt * 32 + lane], Blw[(kt * 8 + wv) * 32 + lane], acc);
      storeC_bias(sh_wout + wv * 16, WOUT_LD, acc, sh_lwb + wv * 16, lane);
    }
    __syncthreads();

    // scores[m,n] = sum_w lv[w]*tanh(wout[m,w] + ue[b0+m,n,w])   (L2 stream)
    for (int pidx = tid; pidx < 16 * NV; pidx += 256) {
      const int m = pidx / NV, n = pidx - m * NV;
      const float* uer = p.ue + ((size_t)(b0 + m) * NV + n) * W_SZ;
      const float* wo  = sh_wout + m * WOUT_LD;
      float s = 0.f;
#pragma unroll 4
      for (int w = 0; w < W_SZ; ++w) s += sh_lv[w] * ftanh(wo[w] + uer[w]);
      sh_sc[m * NV + n] = s;
    }
    __syncthreads();

    // softmax over n, per batch row; wave wv owns rows 2wv, 2wv+1
#pragma unroll
    for (int rr = 0; rr < 2; ++rr) {
      const int m = wv * 2 + rr;
      float* row = sh_sc + m * NV;
      const float a0 = row[lane], a1 = row[lane + 32], a2 = row[lane + 64];
      const float mx = wave_max(fmaxf(a0, fmaxf(a1, a2)));
      const float e0 = __expf(a0 - mx), e1 = __expf(a1 - mx), e2 = __expf(a2 - mx);
      const float inv = 1.f / wave_sum(e0 + e1 + e2);
      row[lane] = e0 * inv; row[lane + 32] = e1 * inv; row[lane + 64] = e2 * inv;
    }
#if HAVE_TDM
    if (wv == 0) __builtin_amdgcn_s_wait_tensorcnt(0);  // sh_xcol ready
#endif
    __syncthreads();

    // x_tilde = alpha * x[:, :, t], scattered straight into A-fragment order
    for (int pidx = tid; pidx < 16 * NV; pidx += 256) {
      const int m = pidx / NV, n = pidx - m * NV;
#if HAVE_TDM
      const float xv = sh_xcol[pidx];
#else
      const float xv = p.x[((size_t)(b0 + m) * NV + n) * W_SZ + t];
#endif
      packA_elem(sh_xtA, m, n, sh_sc[m * NV + n] * xv);
    }
    __syncthreads();

    // gates(16x256) = xt @ Wih^T + h @ Whh^T + (bih+bhh); 16 N-tiles / 8 waves
    for (int nt = wv; nt < 16; nt += 8) {
      v8f acc = {0.f, 0.f, 0.f, 0.f, 0.f, 0.f, 0.f, 0.f};
#pragma unroll
      for (int kt = 0; kt < 3; ++kt)               // K = NV = 96
        acc = wmma_bf16(xtA[kt * 32 + lane], BWih[(kt * 16 + nt) * 32 + lane], acc);
#pragma unroll
      for (int kt = 0; kt < 2; ++kt)               // K = E = 64 (h part of hcA)
        acc = wmma_bf16(hcA[kt * 32 + lane], BWhh[(kt * 16 + nt) * 32 + lane], acc);
      storeC_bias(sh_g + nt * 16, GATE_LD, acc, sh_b256e + nt * 16, lane);
    }
    __syncthreads();

    // LSTM pointwise update; refresh fp32 state + bf16 A-fragments + h_seq
    for (int idx = tid; idx < 1024; idx += 256) {
      const int m = idx >> 6, e = idx & 63;
      const float gi = sh_g[m * GATE_LD + e];
      const float gf = sh_g[m * GATE_LD + 64 + e];
      const float gg = sh_g[m * GATE_LD + 128 + e];
      const float go = sh_g[m * GATE_LD + 192 + e];
      const float c = fsig(gf) * sh_hc[m * HC_LD + 64 + e] + fsig(gi) * ftanh(gg);
      const float h = fsig(go) * ftanh(c);
      sh_hc[m * HC_LD + e] = h;
      sh_hc[m * HC_LD + 64 + e] = c;
      packA_elem(sh_hcA, m, e, h);
      packA_elem(sh_hcA, m, 64 + e, c);
      p.hseq[((size_t)bg * W_SZ + t) * 1024 + m * 64 + e] = h;
    }
    __syncthreads();
  }
  __threadfence();  // h_seq visible for in-block re-reads
  __syncthreads();

  // ================ u_d[w] = h_seq[w] @ dec_lu^T + lub  (WMMA) =============
  for (int w = wv; w < W_SZ; w += 8) {
    const float* hs  = p.hseq + ((size_t)bg * W_SZ + w) * 1024;  // 16x64
    float*       udw = p.ud   + ((size_t)bg * W_SZ + w) * 1024;
#pragma unroll
    for (int nt = 0; nt < 4; ++nt) {
      v8f acc = {0.f, 0.f, 0.f, 0.f, 0.f, 0.f, 0.f, 0.f};
#pragma unroll
      for (int kt = 0; kt < 2; ++kt) {
        v16bf a = loadA_f32(hs, 64, kt * 32, lane);
        acc = wmma_bf16(a, Bdlu[(kt * 4 + nt) * 32 + lane], acc);
      }
      storeC_bias(udw + nt * 16, 64, acc, p.dec_lub + nt * 16, lane);
    }
  }
  for (int i = tid; i < 16 * HC_LD; i += 256) sh_hc[i] = 0.f;  // reset state
  for (int i = tid; i < 4 * 512; i += 256) sh_hcA[i] = (__bf16)0.f;
  __threadfence();
  __syncthreads();

  const int tgt = *p.target;

  // ======================= decoder: 127 steps ==============================
  for (int s = 0; s < W_SZ - 1; ++s) {
    // w_out(16x64) = [h,c] @ dec_lw^T + lwb ; 4 N-tiles on waves 0..3
    if (wv < 4) {
      v8f acc = {0.f, 0.f, 0.f, 0.f, 0.f, 0.f, 0.f, 0.f};
#pragma unroll
      for (int kt = 0; kt < 4; ++kt)
        acc = wmma_bf16(hcA[kt * 32 + lane], Bdlw[(kt * 4 + wv) * 32 + lane], acc);
      storeC_bias(sh_wout + wv * 16, WOUT_LD, acc, sh_lwbd + wv * 16, lane);
    }
    __syncthreads();

    // v[w,m] = sum_e lvd[e]*tanh(wout[m,e] + ud[w,m,e])
    for (int pidx = tid; pidx < 2048; pidx += 256) {
      const int w = pidx >> 4, m = pidx & 15;
      const float* udr = p.ud + ((size_t)bg * W_SZ + w) * 1024 + m * 64;
      const float* wo  = sh_wout + m * WOUT_LD;
      float sv = 0.f;
#pragma unroll 4
      for (int e = 0; e < 64; ++e) sv += sh_lvd[e] * ftanh(wo[e] + udr[e]);
      sh_sc[w * 16 + m] = sv;
    }
    __syncthreads();

    // softmax over w, per batch row; wave wv owns rows 2wv, 2wv+1
#pragma unroll
    for (int rr = 0; rr < 2; ++rr) {
      const int m = wv * 2 + rr;
      float a[4];
#pragma unroll
      for (int q = 0; q < 4; ++q) a[q] = sh_sc[(lane + 32 * q) * 16 + m];
      float mx = fmaxf(fmaxf(a[0], a[1]), fmaxf(a[2], a[3]));
      mx = wave_max(mx);
      float sum = 0.f;
#pragma unroll
      for (int q = 0; q < 4; ++q) { a[q] = __expf(a[q] - mx); sum += a[q]; }
      const float inv = 1.f / wave_sum(sum);
#pragma unroll
      for (int q = 0; q < 4; ++q) sh_sc[(lane + 32 * q) * 16 + m] = a[q] * inv;
    }
    __syncthreads();

    // context[m,e] = sum_w beta[w,m] * h_seq[w,m,e]   (L2 stream)
    for (int idx = tid; idx < 1024; idx += 256) {
      const int m = idx >> 6, e = idx & 63;
      const float* hb = p.hseq + (size_t)bg * W_SZ * 1024 + m * 64 + e;
      float sv = 0.f;
#pragma unroll 4
      for (int w = 0; w < W_SZ; ++w) sv += sh_sc[w * 16 + m] * hb[(size_t)w * 1024];
      sh_ctx[m * 64 + e] = sv;
    }
    __syncthreads();

    // y_tilde = [y_prev, context] @ dec_lp^T + lpb ; wave-parallel 65-dot
#pragma unroll
    for (int rr = 0; rr < 2; ++rr) {
      const int m = wv * 2 + rr;
      float sv = p.dec_lp[1 + lane] * sh_ctx[m * 64 + lane] +
                 p.dec_lp[1 + lane + 32] * sh_ctx[m * 64 + lane + 32];
      sv = wave_sum(sv);
      if (lane == 0) {
        const float yp = p.x[((size_t)(b0 + m) * NV + tgt) * W_SZ + s];
        sh_yt[m] = sv + p.dec_lp[0] * yp + p.dec_lpb[0];
      }
    }
    __syncthreads();

    // gates = h @ dec_Whh^T + (bih+bhh) (y_tilde*Wih folded in pointwise)
    for (int nt = wv; nt < 16; nt += 8) {
      v8f acc = {0.f, 0.f, 0.f, 0.f, 0.f, 0.f, 0.f, 0.f};
#pragma unroll
      for (int kt = 0; kt < 2; ++kt)               // K = D = 64 (h part of hcA)
        acc = wmma_bf16(hcA[kt * 32 + lane], Bdwhh[(kt * 16 + nt) * 32 + lane], acc);
      storeC_bias(sh_g + nt * 16, GATE_LD, acc, sh_b256d + nt * 16, lane);
    }
    __syncthreads();

    // LSTM pointwise update
    for (int idx = tid; idx < 1024; idx += 256) {
      const int m = idx >> 6, e = idx & 63;
      const float yw = sh_yt[m];
      const float gi = sh_g[m * GATE_LD + e]       + yw * p.dec_Wih[e];
      const float gf = sh_g[m * GATE_LD + 64 + e]  + yw * p.dec_Wih[64 + e];
      const float gg = sh_g[m * GATE_LD + 128 + e] + yw * p.dec_Wih[128 + e];
      const float go = sh_g[m * GATE_LD + 192 + e] + yw * p.dec_Wih[192 + e];
      const float c = fsig(gf) * sh_hc[m * HC_LD + 64 + e] + fsig(gi) * ftanh(gg);
      const float h = fsig(go) * ftanh(c);
      sh_hc[m * HC_LD + e] = h;
      sh_hc[m * HC_LD + 64 + e] = c;
      packA_elem(sh_hcA, m, e, h);
      packA_elem(sh_hcA, m, 64 + e, c);
    }
    __syncthreads();
  }

  // out[b] = [h,c] @ dec_ly^T + lyb ; wave-parallel 128-dot, 2 rows per wave
#pragma unroll
  for (int rr = 0; rr < 2; ++rr) {
    const int m = wv * 2 + rr;
    float sv = 0.f;
#pragma unroll
    for (int q = 0; q < 4; ++q)
      sv += sh_hc[m * HC_LD + lane + 32 * q] * p.dec_ly[lane + 32 * q];
    sv = wave_sum(sv);
    if (lane == 0) p.out[b0 + m] = sv + p.dec_lyb[0];
  }
}

// ---------------------------------------------------------------------------
extern "C" void kernel_launch(void* const* d_in, const int* in_sizes, int n_in,
                              void* d_out, int out_size, void* d_ws,
                              size_t ws_size, hipStream_t stream) {
  const float* x       = (const float*)d_in[0];
  const float* enc_Wih = (const float*)d_in[1];
  const float* enc_Whh = (const float*)d_in[2];
  const float* enc_bih = (const float*)d_in[3];
  const float* enc_bhh = (const float*)d_in[4];
  const float* enc_lv  = (const float*)d_in[5];
  const float* enc_lw  = (const float*)d_in[6];
  const float* enc_lwb = (const float*)d_in[7];
  const float* enc_lu  = (const float*)d_in[8];
  const float* enc_lub = (const float*)d_in[9];
  const float* dec_Wih = (const float*)d_in[10];
  const float* dec_Whh = (const float*)d_in[11];
  const float* dec_bih = (const float*)d_in[12];
  const float* dec_bhh = (const float*)d_in[13];
  const float* dec_lv  = (const float*)d_in[14];
  const float* dec_lw  = (const float*)d_in[15];
  const float* dec_lwb = (const float*)d_in[16];
  const float* dec_lu  = (const float*)d_in[17];
  const float* dec_lub = (const float*)d_in[18];
  const float* dec_lp  = (const float*)d_in[19];
  const float* dec_lpb = (const float*)d_in[20];
  const float* dec_ly  = (const float*)d_in[21];
  const float* dec_lyb = (const float*)d_in[22];
  const int*   target  = (const int*)d_in[23];
  (void)in_sizes; (void)n_in; (void)out_size; (void)ws_size;

  char* ws = (char*)d_ws;
  size_t off = 0;
  auto carve = [&](size_t bytes) -> void* {
    void* pp = ws + off;
    off += (bytes + 255) & ~(size_t)255;
    return pp;
  };
  __bf16* Blw   = (__bf16*)carve((size_t)128 * 128 * 2);  // enc_lw^T frags
  __bf16* BWih  = (__bf16*)carve((size_t)96  * 256 * 2);
  __bf16* BWhh  = (__bf16*)carve((size_t)64  * 256 * 2);
  __bf16* Blu   = (__bf16*)carve((size_t)128 * 128 * 2);
  __bf16* Bdlw  = (__bf16*)carve((size_t)128 * 64  * 2);
  __bf16* Bdlu  = (__bf16*)carve((size_t)64  * 64  * 2);
  __bf16* Bdwhh = (__bf16*)carve((size_t)64  * 256 * 2);
  float* ue   = (float*)carve((size_t)B_SZ * NV   * W_SZ * 4);  // 50 MB
  float* hseq = (float*)carve((size_t)B_SZ * W_SZ * E_SZ * 4);  // 32 MB
  float* ud   = (float*)carve((size_t)B_SZ * W_SZ * D_SZ * 4);  // 32 MB

  // Pack all GEMM weights into bf16 WMMA B-fragment order (K/32 x N/16 tiles).
  pack_b<<<dim3((128 / 32) * (128 / 16)), dim3(32), 0, stream>>>(enc_lw, Blw, 128, 128);
  pack_b<<<dim3((96  / 32) * (256 / 16)), dim3(32), 0, stream>>>(enc_Wih, BWih, 96, 256);
  pack_b<<<dim3((64  / 32) * (256 / 16)), dim3(32), 0, stream>>>(enc_Whh, BWhh, 64, 256);
  pack_b<<<dim3((128 / 32) * (128 / 16)), dim3(32), 0, stream>>>(enc_lu, Blu, 128, 128);
  pack_b<<<dim3((128 / 32) * (64  / 16)), dim3(32), 0, stream>>>(dec_lw, Bdlw, 128, 64);
  pack_b<<<dim3((64  / 32) * (64  / 16)), dim3(32), 0, stream>>>(dec_lu, Bdlu, 64, 64);
  pack_b<<<dim3((64  / 32) * (256 / 16)), dim3(32), 0, stream>>>(dec_Whh, Bdwhh, 64, 256);

  // u_e precompute (WMMA), stays L2-resident for the encoder loop.
  ue_kernel<<<dim3(B_SZ, NV / 16), dim3(256), 0, stream>>>(x, Blu, enc_lub, ue);

  DP p;
  p.x = x; p.enc_lv = enc_lv; p.enc_lwb = enc_lwb;
  p.enc_bih = enc_bih; p.enc_bhh = enc_bhh;
  p.dec_Wih = dec_Wih; p.dec_bih = dec_bih; p.dec_bhh = dec_bhh;
  p.dec_lv = dec_lv; p.dec_lwb = dec_lwb; p.dec_lub = dec_lub;
  p.dec_lp = dec_lp; p.dec_lpb = dec_lpb; p.dec_ly = dec_ly; p.dec_lyb = dec_lyb;
  p.target = target;
  p.Blw = Blw; p.BWih = BWih; p.BWhh = BWhh;
  p.Bdlw = Bdlw; p.Bdlu = Bdlu; p.Bdwhh = Bdwhh;
  p.ue = ue; p.hseq = hseq; p.ud = ud; p.out = (float*)d_out;

  darnn_main<<<dim3(B_SZ / 16), dim3(256), 0, stream>>>(p);
}
